// PositionEncoder_30786325578180
// MI455X (gfx1250) — matrix operand order repaired and checked by reading, not compiled, not run on hardware
//
#include <hip/hip_runtime.h>
#include <hip/hip_bf16.h>
#include <math.h>

// ---------------------------------------------------------------------------
// Problem dimensions (fixed by the reference)
// ---------------------------------------------------------------------------
#define NQc     10000
#define NCc     30000
#define KNN     16
#define NEc     (NQc * KNN)
#define FSCA    256
#define FVEC    64
#define ECH     64
#define OSCA    256
#define OVEC    64
#define LEAK    0.2f
#define VN_EPS  1e-6f

#define GS_STEP  (10.0f / 63.0f)
#define GS_COEFF (-0.5f / (GS_STEP * GS_STEP))

typedef __attribute__((ext_vector_type(16))) __bf16 v16bf;
typedef __attribute__((ext_vector_type(8)))  float  v8f;

// ---------------------------------------------------------------------------
// WMMA fragment helpers (wave32, bf16 A/B, f32 accum). Layouts per CDNA5 ISA
// 7.12.2: A 16x32 bf16 (lane m=l&15, K split 8+8 / +16), B 32x16 (lane n=l&15,
// K split 8+8 / +16 as A), C/D f32 (lane n=l&15, M=(l>=16?8:0)+r).
// ---------------------------------------------------------------------------
__device__ __forceinline__ v8f wmma_bf16(v16bf a, v16bf b, v8f c) {
  return __builtin_amdgcn_wmma_f32_16x16x32_bf16(false, a, false, b, (short)0, c,
                                                 false, false);
}

// A fragment from row-major f32: element (m, k0+k) at p[m*rstride + (k0+k)*kstride]
__device__ __forceinline__ v16bf packA(const float* p, int rstride, int kstride,
                                       int lane) {
  const int m  = lane & 15;
  const int kb = (lane >> 4) << 3;
  const float* r = p + m * rstride;
  v16bf a;
#pragma unroll
  for (int j = 0; j < 8; ++j) {
    a[j]     = (__bf16)r[(kb + j) * kstride];
    a[j + 8] = (__bf16)r[(kb + 16 + j) * kstride];
  }
  return a;
}

// B fragment = W^T tile: B[k,n] = W[n,k], W row-major (N x ldw)
__device__ __forceinline__ v16bf packB(const float* W, int ldw, int n0, int k0,
                                       int lane) {
  const float* r = W + (n0 + (lane & 15)) * ldw + k0 + ((lane >> 4) << 3);
  v16bf b;
#pragma unroll
  for (int e = 0; e < 16; ++e) b[e] = (__bf16)r[e < 8 ? e : e + 8];
  return b;
}

__device__ __forceinline__ void storeD(float* dst, int rstride, int n0, int lane,
                                       v8f d) {
  const int n  = n0 + (lane & 15);
  const int mb = (lane >> 4) << 3;
#pragma unroll
  for (int r = 0; r < 8; ++r) dst[(mb + r) * rstride + n] = d[r];
}

// out(16xN) = A(16xK) @ W(NxK)^T (+bias)
__device__ __forceinline__ void gemm_rm(const float* A, int lda, int K,
                                        const float* W, int N, const float* bias,
                                        float* out, int outLd, int lane) {
  for (int n0 = 0; n0 < N; n0 += 16) {
    v8f acc = {};
    for (int k0 = 0; k0 < K; k0 += 32)
      acc = wmma_bf16(packA(A + k0, lda, 1, lane), packB(W, K, n0, k0, lane), acc);
    if (bias) {
      const float bb = bias[n0 + (lane & 15)];
#pragma unroll
      for (int r = 0; r < 8; ++r) acc[r] += bb;
    }
    storeD(out, outLd, n0, lane, acc);
  }
}

// out(16xN) = [A1(16xK1) | A2(16xK2)] @ W(N x (K1+K2))^T   (the GVLinear ws step)
__device__ __forceinline__ void gemm_cat(const float* A1, int ld1, int K1,
                                         const float* A2, int ld2, int K2,
                                         const float* W, int N,
                                         float* out, int outLd, int lane) {
  const int KT = K1 + K2;
  for (int n0 = 0; n0 < N; n0 += 16) {
    v8f acc = {};
    for (int k0 = 0; k0 < K1; k0 += 32)
      acc = wmma_bf16(packA(A1 + k0, ld1, 1, lane), packB(W, KT, n0, k0, lane), acc);
    for (int k0 = 0; k0 < K2; k0 += 32)
      acc = wmma_bf16(packA(A2 + k0, ld2, 1, lane), packB(W, KT, n0, K1 + k0, lane), acc);
    storeD(out, outLd, n0, lane, acc);
  }
}

// Per-spatial-dim GEMM on (16, K, 3) layout: out[m,n,d] = sum_k A[m,k,d]*W[n,k],
// optional elementwise gate(16xN).
__device__ __forceinline__ void gemm_v3(const float* A, int K, const float* W,
                                        int N, float* out, const float* gate,
                                        int lane) {
  for (int d = 0; d < 3; ++d)
    for (int n0 = 0; n0 < N; n0 += 16) {
      v8f acc = {};
      for (int k0 = 0; k0 < K; k0 += 32)
        acc = wmma_bf16(packA(A + d + k0 * 3, K * 3, 3, lane),
                        packB(W, K, n0, k0, lane), acc);
      const int n  = n0 + (lane & 15);
      const int mb = (lane >> 4) << 3;
#pragma unroll
      for (int r = 0; r < 8; ++r) {
        float v = acc[r];
        if (gate) v *= gate[(mb + r) * N + n];
        out[(mb + r) * (N * 3) + n * 3 + d] = v;
      }
    }
}

__device__ __forceinline__ void vnorm16(const float* inter, int K, float* out,
                                        int lane) {
  for (int i = lane; i < 16 * K; i += 32) {
    const float* p = inter + i * 3;  // (m*K+c)*3
    out[i] = sqrtf(p[0] * p[0] + p[1] * p[1] + p[2] * p[2] + 1e-12f);
  }
}

__device__ __forceinline__ void sigmoid16(float* g, int N, int lane) {
  for (int i = lane; i < 16 * N; i += 32) g[i] = 1.f / (1.f + expf(-g[i]));
}

// ---------------------------------------------------------------------------
// Kernel 1: node_gv GVLinear over all NC complex nodes -> ws.ns / ws.nv
// ---------------------------------------------------------------------------
__global__ void k_node_gv(const float* __restrict__ node_sca,
                          const float* __restrict__ node_vec,
                          const float* wv1, const float* wv2, const float* wS,
                          const float* wg, const float* bg,
                          float* __restrict__ ns_out, float* __restrict__ nv_out) {
  __shared__ float sInter[16 * 64 * 3];
  __shared__ float sVn[16 * 64];
  __shared__ float sSca[16 * 256];
  __shared__ float sGate[16 * 64];
  const int lane = threadIdx.x;
  const int m0   = blockIdx.x * 16;

  gemm_v3(node_vec + m0 * 192, 64, wv1, 64, sInter, nullptr, lane);
  __syncthreads();
  vnorm16(sInter, 64, sVn, lane);
  __syncthreads();
  gemm_cat(sVn, 64, 64, node_sca + m0 * 256, 256, 256, wS, 256, sSca, 256, lane);
  __syncthreads();
  for (int i = lane; i < 16 * 256; i += 32) ns_out[m0 * 256 + i] = sSca[i];
  gemm_rm(sSca, 256, 256, wg, 64, bg, sGate, 64, lane);
  __syncthreads();
  sigmoid16(sGate, 64, lane);
  __syncthreads();
  gemm_v3(sInter, 64, wv2, 64, nv_out + m0 * 192, sGate, lane);
}

// ---------------------------------------------------------------------------
// Kernel 2: root branch (root_gv) + att_q scalar head -> ws.qs
// ---------------------------------------------------------------------------
__global__ void k_root_attq(const float* __restrict__ pos_query,
                            const float* __restrict__ atom_emb,
                            const float* root_exp_w,
                            const float* r_wv1, const float* r_wv2,
                            const float* r_ws, const float* r_wg, const float* r_bg,
                            const float* q_wv1, const float* q_ws,
                            float* __restrict__ qs_out) {
  __shared__ float sVec[16 * 64 * 3];
  __shared__ float sInter[16 * 64 * 3];
  __shared__ float sVn[16 * 64];
  __shared__ float sXs[16 * 256];
  __shared__ float sGate[16 * 64];
  __shared__ float sInv[16];
  const int lane = threadIdx.x;
  const int m0   = blockIdx.x * 16;

  if (lane < 16) {
    float x = pos_query[(m0 + lane) * 3 + 0];
    float y = pos_query[(m0 + lane) * 3 + 1];
    float z = pos_query[(m0 + lane) * 3 + 2];
    sInv[lane] = 1.f / (sqrtf(x * x + y * y + z * z + 1e-12f) + 1e-7f);
  }
  __syncthreads();
  for (int i = lane; i < 16 * 64 * 3; i += 32) {
    int m = i / 192, r = i % 192, c = r / 3, d = r % 3;
    sVec[i] = pos_query[(m0 + m) * 3 + d] * sInv[m] * root_exp_w[c];
  }
  __syncthreads();
  // root_gv
  gemm_v3(sVec, 64, r_wv1, 64, sInter, nullptr, lane);  __syncthreads();
  vnorm16(sInter, 64, sVn, lane);                       __syncthreads();
  gemm_cat(sVn, 64, 64, atom_emb + m0 * 256, 256, 256, r_ws, 256, sXs, 256, lane);
  __syncthreads();
  gemm_rm(sXs, 256, 256, r_wg, 64, r_bg, sGate, 64, lane); __syncthreads();
  sigmoid16(sGate, 64, lane);                              __syncthreads();
  gemm_v3(sInter, 64, r_wv2, 64, sVec, sGate, lane);       __syncthreads();  // x_vec
  // att_q scalar path only (its vector output is discarded by the reference)
  gemm_v3(sVec, 64, q_wv1, 64, sInter, nullptr, lane);     __syncthreads();
  vnorm16(sInter, 64, sVn, lane);                          __syncthreads();
  gemm_cat(sVn, 64, 64, sXs, 256, 256, q_ws, 256, qs_out + m0 * 256, 256, lane);
}

// ---------------------------------------------------------------------------
// Kernel 3: fully-fused per-query edge pipeline + attention.
// One block (1 wave32) per query: its 16 contiguous edges are one WMMA M-tile.
// ---------------------------------------------------------------------------
struct EdgeArgs {
  const float *pos_query, *cpx_pos;
  const int*   idx_c;
  const float* exp_w;
  const float *e_wv1, *e_wv2, *e_ws, *e_wg, *e_bg;
  const float* vn_dir;
  const float *sca_w, *sca_b, *e2n_w, *e2n_b, *n2e_w, *n2e_b, *vnlin;
  const float *o_wv1, *o_wv2, *o_ws, *o_wg, *o_bg;
  const float *k_wv1, *k_ws;
  const float *v_wv1, *v_wv2, *v_ws, *v_wg, *v_bg;
  const float *ns, *nv, *qs;
  float *out_sca, *out_vec;
};

__global__ void k_edge(EdgeArgs A) {
  __shared__ float sV3a[16 * 64 * 3];   // inter scratch
  __shared__ float sV3b[16 * 64 * 3];   // edge_vec -> ev -> h_vec
  __shared__ float sV3c[16 * 64 * 3];   // dd -> ev2/y_vec -> vv
  __shared__ float sS256a[16 * 256];    // ns (gathered)
  __shared__ float sS256b[16 * 256];    // y_sca -> ks -> vs
  __shared__ float sS256c[16 * 256];    // h_sca
  __shared__ float sS64a[16 * 64];      // edge_sca
  __shared__ float sS64b[16 * 64];      // vnorm / gate_e / t2
  __shared__ float sS64c[16 * 64];      // es
  __shared__ float sS64d[16 * 64];      // t1 / gate_h / gate_v
  __shared__ float sDir[16 * 4];        // nx,ny,nz,dist
  __shared__ int   sIdx[16];
  __shared__ float sLog[16 * 4];
  const int lane = threadIdx.x;
  const int q    = blockIdx.x;

  // --- edge geometry -------------------------------------------------------
  if (lane < 16) {
    const int e = q * KNN + lane;
    const int c = A.idx_c[e];
    sIdx[lane]  = c;
    float dx = A.pos_query[q * 3 + 0] - A.cpx_pos[c * 3 + 0];
    float dy = A.pos_query[q * 3 + 1] - A.cpx_pos[c * 3 + 1];
    float dz = A.pos_query[q * 3 + 2] - A.cpx_pos[c * 3 + 2];
    float dist = sqrtf(dx * dx + dy * dy + dz * dz + 1e-12f);
    float inv  = 1.f / (dist + 1e-7f);
    sDir[lane * 4 + 0] = dx * inv;
    sDir[lane * 4 + 1] = dy * inv;
    sDir[lane * 4 + 2] = dz * inv;
    sDir[lane * 4 + 3] = dist;
  }
  __syncthreads();
  for (int i = lane; i < 16 * 64; i += 32) {                 // Gaussian smearing
    int m = i >> 6, k = i & 63;
    float df = sDir[m * 4 + 3] - (float)k * GS_STEP;
    sS64a[i] = expf(GS_COEFF * df * df);
  }
  for (int i = lane; i < 16 * 64 * 3; i += 32) {             // edge_vec
    int m = i / 192, r = i % 192, c = r / 3, d = r % 3;
    sV3b[i] = sDir[m * 4 + d] * A.exp_w[c];
  }
  __syncthreads();

  // --- edge_gv GVLinear ----------------------------------------------------
  gemm_v3(sV3b, 64, A.e_wv1, 64, sV3a, nullptr, lane);  __syncthreads();
  vnorm16(sV3a, 64, sS64b, lane);                        __syncthreads();
  gemm_cat(sS64b, 64, 64, sS64a, 64, 64, A.e_ws, 64, sS64c, 64, lane);
  __syncthreads();
  gemm_rm(sS64c, 64, 64, A.e_wg, 64, A.e_bg, sS64b, 64, lane); __syncthreads();
  sigmoid16(sS64b, 64, lane);                                   __syncthreads();
  gemm_v3(sV3a, 64, A.e_wv2, 64, sV3b, sS64b, lane);            __syncthreads(); // ev

  // --- VN leaky relu on ev, leaky relu on es -------------------------------
  gemm_v3(sV3b, 64, A.vn_dir, 64, sV3c, nullptr, lane);  __syncthreads();
  for (int i = lane; i < 16 * 64; i += 32) {
    float* x        = &sV3b[i * 3];
    const float* dd = &sV3c[i * 3];
    float dot = x[0] * dd[0] + x[1] * dd[1] + x[2] * dd[2];
    float dsq = dd[0] * dd[0] + dd[1] * dd[1] + dd[2] * dd[2];
    float f   = dot / (dsq + VN_EPS);
    bool  pos = dot >= 0.f;
#pragma unroll
    for (int d = 0; d < 3; ++d) {
      float neg = x[d] - f * dd[d];
      x[d] = LEAK * x[d] + (1.f - LEAK) * (pos ? x[d] : neg);
    }
  }
  for (int i = lane; i < 16 * 64; i += 32) {
    float v = sS64c[i];
    sS64c[i] = v >= 0.f ? v : LEAK * v;
  }
  for (int i = lane; i < 16 * 256; i += 32)                // gather ns
    sS256a[i] = A.ns[sIdx[i >> 8] * 256 + (i & 255)];
  __syncthreads();

  // --- elementwise mixing --------------------------------------------------
  gemm_rm(sS64c, 64, 64, A.sca_w, 256, A.sca_b, sS256b, 256, lane);
  __syncthreads();
  for (int i = lane; i < 16 * 256; i += 32) sS256b[i] *= sS256a[i];   // y_sca
  gemm_rm(sS64c, 64, 64, A.e2n_w, 64, A.e2n_b, sS64d, 64, lane);      // t1
  gemm_rm(sS256a, 256, 256, A.n2e_w, 64, A.n2e_b, sS64b, 64, lane);   // t2
  gemm_v3(sV3b, 64, A.vnlin, 64, sV3c, nullptr, lane);                // ev2
  __syncthreads();
  for (int i = lane; i < 16 * 64 * 3; i += 32) {                      // y_vec
    int m = i / 192, r = i % 192, o = r / 3;
    float nvv = A.nv[sIdx[m] * 192 + r];
    sV3c[i] = sS64d[m * 64 + o] * nvv + sS64b[m * 64 + o] * sV3c[i];
  }
  __syncthreads();

  // --- out_gv GVLinear -> per-edge messages h ------------------------------
  gemm_v3(sV3c, 64, A.o_wv1, 64, sV3a, nullptr, lane);  __syncthreads();
  vnorm16(sV3a, 64, sS64b, lane);                        __syncthreads();
  gemm_cat(sS64b, 64, 64, sS256b, 256, 256, A.o_ws, 256, sS256c, 256, lane);
  __syncthreads();
  gemm_rm(sS256c, 256, 256, A.o_wg, 64, A.o_bg, sS64d, 64, lane); __syncthreads();
  sigmoid16(sS64d, 64, lane);                                      __syncthreads();
  gemm_v3(sV3a, 64, A.o_wv2, 64, sV3b, sS64d, lane);               __syncthreads(); // h_vec

  // --- att_k (scalar path) + logits ----------------------------------------
  gemm_v3(sV3b, 64, A.k_wv1, 64, sV3a, nullptr, lane);  __syncthreads();
  vnorm16(sV3a, 64, sS64b, lane);                        __syncthreads();
  gemm_cat(sS64b, 64, 64, sS256c, 256, 256, A.k_ws, 256, sS256b, 256, lane);
  __syncthreads();
  {
    const float* qrow = A.qs + q * 256;
    const int m  = lane & 15;
    const int h0 = (lane >> 4) * 2;
    for (int t = 0; t < 2; ++t) {
      const int h = h0 + t;
      float acc = 0.f;
      for (int j = 0; j < 64; ++j)
        acc += qrow[h * 64 + j] * sS256b[m * 256 + h * 64 + j];
      sLog[m * 4 + h] = acc * 0.125f;   // 1/sqrt(64)
    }
  }
  __syncthreads();
  if (lane < 4) {                        // per-head softmax over the 16 edges
    const int h = lane;
    float mx = -1e30f;
    for (int m = 0; m < 16; ++m) mx = fmaxf(mx, sLog[m * 4 + h]);
    float s = 0.f;
    for (int m = 0; m < 16; ++m) {
      float e = expf(sLog[m * 4 + h] - mx);
      sLog[m * 4 + h] = e;
      s += e;
    }
    float inv = 1.f / s;
    for (int m = 0; m < 16; ++m) sLog[m * 4 + h] *= inv;
  }
  __syncthreads();

  // --- att_v (full GVLinear) -----------------------------------------------
  gemm_v3(sV3b, 64, A.v_wv1, 64, sV3a, nullptr, lane);  __syncthreads();
  vnorm16(sV3a, 64, sS64b, lane);                        __syncthreads();
  gemm_cat(sS64b, 64, 64, sS256c, 256, 256, A.v_ws, 256, sS256b, 256, lane);
  __syncthreads();
  gemm_rm(sS256b, 256, 256, A.v_wg, 64, A.v_bg, sS64d, 64, lane); __syncthreads();
  sigmoid16(sS64d, 64, lane);                                      __syncthreads();
  gemm_v3(sV3a, 64, A.v_wv2, 64, sV3c, sS64d, lane);               __syncthreads(); // vv

  // --- attention-weighted reduction -> output ------------------------------
  for (int i = lane; i < 256; i += 32) {
    const int h = i >> 6;
    float acc = 0.f;
    for (int m = 0; m < 16; ++m) acc += sLog[m * 4 + h] * sS256b[m * 256 + i];
    A.out_sca[q * 256 + i] = acc;
  }
  for (int i = lane; i < 192; i += 32) {
    const int h = (i / 3) >> 4;
    float acc = 0.f;
    for (int m = 0; m < 16; ++m) acc += sLog[m * 4 + h] * sV3c[m * 192 + i];
    A.out_vec[q * 192 + i] = acc;
  }
}

// ---------------------------------------------------------------------------
// Host launcher
// ---------------------------------------------------------------------------
extern "C" void kernel_launch(void* const* d_in, const int* in_sizes, int n_in,
                              void* d_out, int out_size, void* d_ws, size_t ws_size,
                              hipStream_t stream) {
  (void)in_sizes; (void)n_in; (void)out_size; (void)ws_size;
  const float* pos_query  = (const float*)d_in[0];
  const int*   edge_index = (const int*)d_in[1];
  const float* cpx_pos    = (const float*)d_in[2];
  const float* node_sca   = (const float*)d_in[3];
  const float* node_vec   = (const float*)d_in[4];
  const float* atom_emb   = (const float*)d_in[5];
  const float* exp_w      = (const float*)d_in[6];
  const float* root_exp_w = (const float*)d_in[7];
  // GV param blocks in insertion order: wv1, wv2, ws, wg, bg
  const float *r_wv1 = (const float*)d_in[8],  *r_wv2 = (const float*)d_in[9];
  const float *r_ws  = (const float*)d_in[10], *r_wg  = (const float*)d_in[11];
  const float *r_bg  = (const float*)d_in[12];
  const float *n_wv1 = (const float*)d_in[13], *n_wv2 = (const float*)d_in[14];
  const float *n_ws  = (const float*)d_in[15], *n_wg  = (const float*)d_in[16];
  const float *n_bg  = (const float*)d_in[17];
  const float *e_wv1 = (const float*)d_in[18], *e_wv2 = (const float*)d_in[19];
  const float *e_ws  = (const float*)d_in[20], *e_wg  = (const float*)d_in[21];
  const float *e_bg  = (const float*)d_in[22];
  const float *vn_dir = (const float*)d_in[23];
  const float *sca_w = (const float*)d_in[24], *sca_b = (const float*)d_in[25];
  const float *e2n_w = (const float*)d_in[26], *e2n_b = (const float*)d_in[27];
  const float *n2e_w = (const float*)d_in[28], *n2e_b = (const float*)d_in[29];
  const float *vnlin = (const float*)d_in[30];
  const float *o_wv1 = (const float*)d_in[31], *o_wv2 = (const float*)d_in[32];
  const float *o_ws  = (const float*)d_in[33], *o_wg  = (const float*)d_in[34];
  const float *o_bg  = (const float*)d_in[35];
  const float *q_wv1 = (const float*)d_in[36], *q_ws  = (const float*)d_in[38];
  const float *k_wv1 = (const float*)d_in[41], *k_ws  = (const float*)d_in[43];
  const float *v_wv1 = (const float*)d_in[46], *v_wv2 = (const float*)d_in[47];
  const float *v_ws  = (const float*)d_in[48], *v_wg  = (const float*)d_in[49];
  const float *v_bg  = (const float*)d_in[50];

  const int* idx_c = edge_index + NEc;   // row 1 of (2, E)

  float* ws_ns = (float*)d_ws;                 // NC x 256
  float* ws_nv = ws_ns + (size_t)NCc * 256;    // NC x 64 x 3
  float* ws_qs = ws_nv + (size_t)NCc * 192;    // NQ x 256

  float* out_sca = (float*)d_out;              // NQ x 256
  float* out_vec = out_sca + (size_t)NQc * 256;// NQ x 64 x 3

  k_node_gv<<<NCc / 16, 32, 0, stream>>>(node_sca, node_vec, n_wv1, n_wv2, n_ws,
                                         n_wg, n_bg, ws_ns, ws_nv);
  k_root_attq<<<NQc / 16, 32, 0, stream>>>(pos_query, atom_emb, root_exp_w,
                                           r_wv1, r_wv2, r_ws, r_wg, r_bg,
                                           q_wv1, q_ws, ws_qs);
  EdgeArgs EA;
  EA.pos_query = pos_query; EA.cpx_pos = cpx_pos; EA.idx_c = idx_c;
  EA.exp_w = exp_w;
  EA.e_wv1 = e_wv1; EA.e_wv2 = e_wv2; EA.e_ws = e_ws; EA.e_wg = e_wg; EA.e_bg = e_bg;
  EA.vn_dir = vn_dir;
  EA.sca_w = sca_w; EA.sca_b = sca_b; EA.e2n_w = e2n_w; EA.e2n_b = e2n_b;
  EA.n2e_w = n2e_w; EA.n2e_b = n2e_b; EA.vnlin = vnlin;
  EA.o_wv1 = o_wv1; EA.o_wv2 = o_wv2; EA.o_ws = o_ws; EA.o_wg = o_wg; EA.o_bg = o_bg;
  EA.k_wv1 = k_wv1; EA.k_ws = k_ws;
  EA.v_wv1 = v_wv1; EA.v_wv2 = v_wv2; EA.v_ws = v_ws; EA.v_wg = v_wg; EA.v_bg = v_bg;
  EA.ns = ws_ns; EA.nv = ws_nv; EA.qs = ws_qs;
  EA.out_sca = out_sca; EA.out_vec = out_vec;
  k_edge<<<NQc, 32, 0, stream>>>(EA);
}
